// CausalRepurposingNet_20847771255408
// MI455X (gfx1250) — compile-verified, matrix operand order checked by reading, not run on hardware
//
#include <hip/hip_runtime.h>
#include <hip/hip_bf16.h>
#include <math.h>

// Problem constants (match reference)
#define D 256
#define N_DRUG 20000
#define N_GENE 20000
#define N_DIS  10000
#define N_TOT  50000

typedef __attribute__((ext_vector_type(16))) __bf16 v16bf;
typedef __attribute__((ext_vector_type(8)))  float  v8f;

union AFrag { v16bf v; unsigned u[8]; };

// ---------------------------------------------------------------------------
// CDNA5 async global->LDS copy (16 bytes per lane, ASYNCcnt-tracked)
// ---------------------------------------------------------------------------
__device__ __forceinline__ void async_copy16(const __bf16* g, void* l) {
  unsigned lds = (unsigned)(uintptr_t)l;            // low 32 bits = LDS offset
  unsigned long long ga = (unsigned long long)(uintptr_t)g;
  asm volatile("global_load_async_to_lds_b128 %0, %1, off"
               :: "v"(lds), "v"(ga) : "memory");
}
__device__ __forceinline__ void async_wait0() {
  asm volatile("s_wait_asynccnt 0" ::: "memory");
}

// Assemble a 16-bit WMMA fragment from an LDS tile stored [rc][k] with 32
// bf16 (16 dwords) per rc. Layout per ISA 7.12.2: lanes 0-15 = rc 0-15,
// VGPR p<4 -> K = 2p + 8*half, p>=4 -> K = 16 + 2(p-4) + 8*half.
__device__ __forceinline__ void load_frag16(const unsigned* __restrict__ base,
                                            int rc, int half, AFrag& f) {
#pragma unroll
  for (int p = 0; p < 8; ++p) {
    int k = ((p < 4) ? (2 * p) : (16 + 2 * (p - 4))) + half * 8;
    f.u[p] = base[rc * 16 + (k >> 1)];
  }
}

// ---------------------------------------------------------------------------
// One-time conversions (per launch): features f32 -> bf16 row-major,
// weights f32 row-major -> bf16 transposed [col][k].
// ---------------------------------------------------------------------------
__global__ __launch_bounds__(256) void conv_f32_bf16(
    const float* __restrict__ in, __bf16* __restrict__ out, int n4) {
  int i = blockIdx.x * 256 + threadIdx.x;
  if (i >= n4) return;
  float4 v = ((const float4*)in)[i];
  union { __bf16 h[4]; uint2 u; } p;
  p.h[0] = (__bf16)v.x; p.h[1] = (__bf16)v.y;
  p.h[2] = (__bf16)v.z; p.h[3] = (__bf16)v.w;
  ((uint2*)out)[i] = p.u;
}

__global__ __launch_bounds__(256) void conv_w_bf16t(
    const float* __restrict__ W, __bf16* __restrict__ Wt, int nElem) {
  int i = blockIdx.x * 256 + threadIdx.x;
  if (i >= nElem) return;
  int mat = i >> 16, rem = i & 65535, k = rem >> 8, col = rem & 255;
  Wt[((size_t)mat << 16) + (size_t)col * D + k] = (__bf16)W[i];
}

// ---------------------------------------------------------------------------
// Tile staging: A 64x32 bf16 (rows), B 256x32 bf16 ([col][k]) via async LDS.
// ---------------------------------------------------------------------------
__device__ __forceinline__ void stage_tiles(
    const __bf16* __restrict__ Xb, const __bf16* __restrict__ Wt,
    __bf16* lA, __bf16* lB, int tid, int rowBase, int N, int k0) {
  {
    int r = tid >> 2, c = tid & 3;
    int gr = rowBase + r;
    if (gr >= N) gr = N - 1;                   // clamp; stores are guarded
    async_copy16(Xb + (size_t)gr * D + k0 + c * 8, &lA[r * 32 + c * 8]);
  }
#pragma unroll
  for (int j = 0; j < 4; ++j) {
    int idx = j * 256 + tid;
    int col = idx >> 2, c = idx & 3;
    async_copy16(Wt + (size_t)col * D + k0 + c * 8, &lB[col * 32 + c * 8]);
  }
  async_wait0();
}

// ---------------------------------------------------------------------------
// T = X @ W   (X: N x 256 bf16 row-major, Wt: [col][k] bf16, T: N x 256 f32)
// One workgroup = 64 rows x 256 cols; 8 waves, wave = 16 rows x 128 cols.
// ---------------------------------------------------------------------------
__global__ __launch_bounds__(256) void rel_gemm_bf16(
    const __bf16* __restrict__ X, const __bf16* __restrict__ Wt,
    float* __restrict__ T, int N)
{
  __shared__ __align__(16) __bf16 lA[64 * 32];
  __shared__ __align__(16) __bf16 lB[256 * 32];

  const int tid  = threadIdx.x;
  const int lane = tid & 31;
  const int wave = tid >> 5;
  const int wr   = wave >> 1;
  const int wc   = wave & 1;
  const int half = lane >> 4;
  const int l16  = lane & 15;
  const int rowBase = blockIdx.x * 64;

  v8f acc[8] = {};

  for (int k0 = 0; k0 < D; k0 += 32) {
    stage_tiles(X, Wt, lA, lB, tid, rowBase, N, k0);
    __syncthreads();

    AFrag a;
    load_frag16((const unsigned*)lA, wr * 16 + l16, half, a);
    AFrag b[8];
#pragma unroll
    for (int cb = 0; cb < 8; ++cb)
      load_frag16((const unsigned*)lB, wc * 128 + cb * 16 + l16, half, b[cb]);
#pragma unroll
    for (int cb = 0; cb < 8; ++cb)
      acc[cb] = __builtin_amdgcn_wmma_f32_16x16x32_bf16(
          false, a.v, false, b[cb].v, (short)0, acc[cb], false, false);
    __syncthreads();
  }

  // C/D layout: lanes 0-15 VGPR r -> M=r, lanes 16-31 -> M=8+r; N = l16.
  if (rowBase + 64 <= N) {
#pragma unroll
    for (int cb = 0; cb < 8; ++cb) {
      int col = wc * 128 + cb * 16 + l16;
#pragma unroll
      for (int r = 0; r < 8; ++r) {
        int row = rowBase + wr * 16 + half * 8 + r;
        T[(size_t)row * D + col] = acc[cb][r];
      }
    }
  } else {
#pragma unroll
    for (int cb = 0; cb < 8; ++cb) {
      int col = wc * 128 + cb * 16 + l16;
#pragma unroll
      for (int r = 0; r < 8; ++r) {
        int row = rowBase + wr * 16 + half * 8 + r;
        if (row < N) T[(size_t)row * D + col] = acc[cb][r];
      }
    }
  }
}

// ---------------------------------------------------------------------------
// Edge scatter:  agg[dst] += m * T[src] ; wsum[dst] += m.  One wave per edge.
// ---------------------------------------------------------------------------
__global__ __launch_bounds__(256) void edge_scatter(
    const float* __restrict__ T, const int* __restrict__ src,
    const int* __restrict__ dst, const float* __restrict__ m,
    float* __restrict__ agg, float* __restrict__ wsum, int E)
{
  int e = blockIdx.x * 8 + (threadIdx.x >> 5);
  if (e >= E) return;
  int lane = threadIdx.x & 31;
  int s = src[e];
  int d = dst[e];
  float mv = m[e];
  const float4* trow = (const float4*)(T + (size_t)s * D);
  float* arow = agg + (size_t)d * D;
#pragma unroll
  for (int j = 0; j < 2; ++j) {
    int q = lane + j * 32;
    float4 v = trow[q];
    atomicAdd(arow + q * 4 + 0, v.x * mv);
    atomicAdd(arow + q * 4 + 1, v.y * mv);
    atomicAdd(arow + q * 4 + 2, v.z * mv);
    atomicAdd(arow + q * 4 + 3, v.w * mv);
  }
  if (lane == 0) atomicAdd(wsum + d, mv);
}

// ---------------------------------------------------------------------------
// Fused:  y = GELU(LN(H @ Wr + b + agg / max(wsum,1)))
// storeBF != 0 -> write bf16 (hidden layer), else f32 (final output).
// ---------------------------------------------------------------------------
__global__ __launch_bounds__(256) void combine_ln_gelu(
    const __bf16* __restrict__ H, const __bf16* __restrict__ Wt,
    const float* __restrict__ bias, const float* __restrict__ gamma,
    const float* __restrict__ beta, const float* __restrict__ agg,
    const float* __restrict__ wsum, float* __restrict__ OutF,
    __bf16* __restrict__ OutB, int N, int storeBF)
{
  __shared__ __align__(16) __bf16 lA[64 * 32];
  __shared__ __align__(16) __bf16 lB[256 * 32];
  __shared__ float sSum[64], sSq[64], sMu[64], sRs[64], sInvDen[64];

  const int tid  = threadIdx.x;
  const int lane = tid & 31;
  const int wave = tid >> 5;
  const int wr   = wave >> 1;
  const int wc   = wave & 1;
  const int half = lane >> 4;
  const int l16  = lane & 15;
  const int rowBase = blockIdx.x * 64;

  v8f acc[8] = {};

  for (int k0 = 0; k0 < D; k0 += 32) {
    stage_tiles(H, Wt, lA, lB, tid, rowBase, N, k0);
    __syncthreads();

    AFrag a;
    load_frag16((const unsigned*)lA, wr * 16 + l16, half, a);
    AFrag b[8];
#pragma unroll
    for (int cb = 0; cb < 8; ++cb)
      load_frag16((const unsigned*)lB, wc * 128 + cb * 16 + l16, half, b[cb]);
#pragma unroll
    for (int cb = 0; cb < 8; ++cb)
      acc[cb] = __builtin_amdgcn_wmma_f32_16x16x32_bf16(
          false, a.v, false, b[cb].v, (short)0, acc[cb], false, false);
    __syncthreads();
  }

  const bool full = (rowBase + 64) <= N;

  // ---- epilogue: + bias + agg/den, LN stats via LDS atomics, LN+GELU ----
  if (tid < 64) {
    int grow = rowBase + tid;
    float w = (grow < N) ? wsum[grow] : 1.0f;
    sInvDen[tid] = 1.0f / fmaxf(w, 1.0f);
    sSum[tid] = 0.0f;
    sSq[tid]  = 0.0f;
  }
  __syncthreads();

  float rowPart[8]   = {};
  float rowPartSq[8] = {};
#pragma unroll
  for (int cb = 0; cb < 8; ++cb) {
    int col = wc * 128 + cb * 16 + l16;
    float rb = bias[col];
#pragma unroll
    for (int r = 0; r < 8; ++r) {
      int rBlk = wr * 16 + half * 8 + r;
      int grow = rowBase + rBlk;
      float a = (full || grow < N) ? agg[(size_t)grow * D + col] : 0.0f;
      float v = acc[cb][r] + rb + a * sInvDen[rBlk];
      acc[cb][r] = v;
      rowPart[r]   += v;
      rowPartSq[r] += v * v;
    }
  }
#pragma unroll
  for (int r = 0; r < 8; ++r) {
    int rBlk = wr * 16 + half * 8 + r;
    atomicAdd(&sSum[rBlk], rowPart[r]);     // ds_add_f32
    atomicAdd(&sSq[rBlk],  rowPartSq[r]);
  }
  __syncthreads();

  if (tid < 64) {
    float mu  = sSum[tid] * (1.0f / 256.0f);
    float var = sSq[tid] * (1.0f / 256.0f) - mu * mu;
    sMu[tid] = mu;
    sRs[tid] = rsqrtf(fmaxf(var, 0.0f) + 1e-5f);
  }
  __syncthreads();

#pragma unroll
  for (int cb = 0; cb < 8; ++cb) {
    int col = wc * 128 + cb * 16 + l16;
    float g  = gamma[col];
    float b2 = beta[col];
#pragma unroll
    for (int r = 0; r < 8; ++r) {
      int rBlk = wr * 16 + half * 8 + r;
      int grow = rowBase + rBlk;
      if (full || grow < N) {
        float v = (acc[cb][r] - sMu[rBlk]) * sRs[rBlk] * g + b2;
        float ge = 0.5f * v * (1.0f + erff(v * 0.70710678118654752f));
        if (storeBF) OutB[(size_t)grow * D + col] = (__bf16)ge;
        else         OutF[(size_t)grow * D + col] = ge;
      }
    }
  }
}

// ---------------------------------------------------------------------------
extern "C" void kernel_launch(void* const* d_in, const int* in_sizes, int n_in,
                              void* d_out, int out_size, void* d_ws, size_t ws_size,
                              hipStream_t stream)
{
  const float* x[3]     = {(const float*)d_in[0], (const float*)d_in[1], (const float*)d_in[2]};
  const int*   srcs[4]  = {(const int*)d_in[3], (const int*)d_in[5], (const int*)d_in[7], (const int*)d_in[9]};
  const int*   dsts[4]  = {(const int*)d_in[4], (const int*)d_in[6], (const int*)d_in[8], (const int*)d_in[10]};
  const float* masks[4] = {(const float*)d_in[11], (const float*)d_in[12], (const float*)d_in[13], (const float*)d_in[14]};
  const float* rel_w  = (const float*)d_in[15];
  const float* root_w = (const float*)d_in[16];
  const float* root_b = (const float*)d_in[17];
  const float* ln_g   = (const float*)d_in[18];
  const float* ln_b   = (const float*)d_in[19];
  const int E = in_sizes[3];

  const int nCnt[3]    = {N_DRUG, N_GENE, N_DIS};
  const int nBase[3]   = {0, N_DRUG, N_DRUG + N_GENE};
  const int srcType[4] = {0, 1, 0, 1};                 // drug, gene, drug, gene
  const int dstBase[4] = {N_DRUG, N_DRUG + N_GENE,     // gene, disease,
                          N_DRUG + N_GENE, N_DRUG};    // disease, gene

  // workspace layout:  T | agg | wsum (f32)  ||  XB | HB | WtRel | WtRoot (bf16)
  float*  T      = (float*)d_ws;                        // 20000*256 f32
  float*  agg    = T + (size_t)N_DRUG * D;              // 50000*256 f32
  float*  wsum   = agg + (size_t)N_TOT * D;             // 50000 f32
  __bf16* XB     = (__bf16*)(wsum + N_TOT);             // 50000*256 bf16
  __bf16* HB     = XB + (size_t)N_TOT * D;              // 50000*256 bf16
  __bf16* WtRel  = HB + (size_t)N_TOT * D;              // 8*65536 bf16
  __bf16* WtRoot = WtRel + (size_t)8 * D * D;           // 6*65536 bf16

  // ---- one-time precision/layout conversion ----
  for (int t = 0; t < 3; ++t) {
    int n4 = nCnt[t] * D / 4;
    conv_f32_bf16<<<dim3((n4 + 255) / 256), dim3(256), 0, stream>>>(
        x[t], XB + (size_t)nBase[t] * D, n4);
  }
  conv_w_bf16t<<<dim3(8 * D * D / 256), dim3(256), 0, stream>>>(rel_w, WtRel, 8 * D * D);
  conv_w_bf16t<<<dim3(6 * D * D / 256), dim3(256), 0, stream>>>(root_w, WtRoot, 6 * D * D);

  for (int l = 0; l < 2; ++l) {
    hipMemsetAsync(agg, 0, ((size_t)N_TOT * D + N_TOT) * sizeof(float), stream);

    const __bf16* Hbase = (l == 0) ? XB : HB;

    for (int rid = 0; rid < 4; ++rid) {
      int st = srcType[rid];
      int Ns = nCnt[st];
      rel_gemm_bf16<<<dim3((Ns + 63) / 64), dim3(256), 0, stream>>>(
          Hbase + (size_t)nBase[st] * D,
          WtRel + (size_t)(l * 4 + rid) * D * D, T, Ns);
      edge_scatter<<<dim3((E + 7) / 8), dim3(256), 0, stream>>>(
          T, srcs[rid], dsts[rid], masks[rid],
          agg + (size_t)dstBase[rid] * D, wsum + dstBase[rid], E);
    }
    for (int t = 0; t < 3; ++t) {
      int off = l * 3 + t;
      combine_ln_gelu<<<dim3((nCnt[t] + 63) / 64), dim3(256), 0, stream>>>(
          Hbase + (size_t)nBase[t] * D,
          WtRoot + (size_t)off * D * D, root_b + (size_t)off * D,
          ln_g + (size_t)off * D, ln_b + (size_t)off * D,
          agg + (size_t)nBase[t] * D, wsum + nBase[t],
          (l == 0) ? nullptr : ((float*)d_out + (size_t)nBase[t] * D),
          (l == 0) ? (HB + (size_t)nBase[t] * D) : nullptr,
          nCnt[t], (l == 0) ? 1 : 0);
    }
  }
  (void)n_in; (void)out_size; (void)ws_size;
}
